// SpatialSelfCrossAttention_48292612276563
// MI455X (gfx1250) — matrix operand order repaired and checked by reading, not compiled, not run on hardware
//
#include <hip/hip_runtime.h>

typedef __attribute__((ext_vector_type(16))) _Float16 v16h;
typedef __attribute__((ext_vector_type(2)))  _Float16 h2;
typedef __attribute__((ext_vector_type(8)))  float    v8f;
typedef __attribute__((ext_vector_type(2)))  float    f2;

#define B_  32
#define C_  256
#define HW_ 1024
#define WSZ (C_ * C_)   // 65536 elements per weight matrix

static __device__ __forceinline__ v8f wmma_f16(v16h a, v16h b, v8f c) {
  // D = A(16x32 f16) * B(32x16 f16) + C(16x16 f32)
  return __builtin_amdgcn_wmma_f32_16x16x32_f16(false, a, false, b, (short)0, c, false, false);
}

// -------- kernel W: one-shot fp32 -> f16 conversion of all six weight matrices
__global__ __launch_bounds__(256) void k_wcvt(
    const float* __restrict__ w0, const float* __restrict__ w1,
    const float* __restrict__ w2, const float* __restrict__ w3,
    const float* __restrict__ w4, const float* __restrict__ w5,
    _Float16* __restrict__ out) {
  const int idx  = (blockIdx.x * 256 + threadIdx.x) * 2;   // 2 elems/thread
  const int wsel = idx >> 16;
  const int off  = idx & (WSZ - 1);
  const float* src = (wsel == 0) ? w0 : (wsel == 1) ? w1 : (wsel == 2) ? w2
                   : (wsel == 3) ? w3 : (wsel == 4) ? w4 : w5;
  f2 v = *(const f2*)(src + off);
  h2 o; o[0] = (_Float16)v[0]; o[1] = (_Float16)v[1];
  *(h2*)(out + (size_t)wsel * WSZ + off) = o;
}

// -------- kernel 0: [B,C,HW] fp32 -> token-major [B,HW,C] f16 (tiled transpose+convert)
__global__ __launch_bounds__(256) void k_tokenize(const float* __restrict__ x1,
                                                  const float* __restrict__ x2,
                                                  _Float16* __restrict__ x1t,
                                                  _Float16* __restrict__ x2t) {
  __shared__ float tile[32][33];
  const int z = blockIdx.z;
  const float* in = (z < B_) ? (x1 + (size_t)z * C_ * HW_) : (x2 + (size_t)(z - B_) * C_ * HW_);
  _Float16*   out = (z < B_) ? (x1t + (size_t)z * HW_ * C_) : (x2t + (size_t)(z - B_) * HW_ * C_);
  const int p0 = blockIdx.x * 32, c0 = blockIdx.y * 32;
  const int tx = threadIdx.x & 31, ty = threadIdx.x >> 5;
  #pragma unroll
  for (int i = ty; i < 32; i += 8)
    tile[i][tx] = in[(size_t)(c0 + i) * HW_ + p0 + tx];
  __syncthreads();
  #pragma unroll
  for (int i = ty; i < 32; i += 8)
    out[(size_t)(p0 + i) * C_ + c0 + tx] = (_Float16)tile[tx][i];
}

// -------- kernel 1: fused projections. proj 0=Q1*scale, 1=K, 2=Q2*scale (token-major f16),
//                    3=V stored channel-major [B,C,HW] f16 (so P*V B-operand is contiguous).
__global__ __launch_bounds__(256) void k_proj(
    const _Float16* __restrict__ x1t, const _Float16* __restrict__ x2t,
    const _Float16* __restrict__ wh,                      // 6 pre-converted weights
    const float* __restrict__ bq1, const float* __restrict__ bk,
    const float* __restrict__ bv,  const float* __restrict__ bq2,
    _Float16* __restrict__ q1s, _Float16* __restrict__ ktok,
    _Float16* __restrict__ q2s, _Float16* __restrict__ vt) {
  const int lane = threadIdx.x & 31;
  const int wid  = threadIdx.x >> 5;
  const int lg   = lane >> 4;
  const int n    = lane & 15;
  const int job  = blockIdx.y * 8 + wid;     // 0..63
  const int proj = job >> 4;                 // 0..3
  const int o0   = (job & 15) * 16;
  const int b    = blockIdx.x >> 6;
  const int p0   = (blockIdx.x & 63) * 16;

  const _Float16* xt = (proj == 2) ? x2t : x1t;
  // weight order in wh: wq1, wk, wv, wq2  ->  proj 0,1,3,2
  const int wsel = (proj == 0) ? 0 : (proj == 1) ? 1 : (proj == 2) ? 3 : 2;
  const _Float16* W = wh + (size_t)wsel * WSZ;
  const float* bias = (proj == 0) ? bq1 : (proj == 1) ? bk : (proj == 2) ? bq2 : bv;

  v8f acc = {};
  if (proj != 3) {
    // D[p][o]: A = Xtok tile (16 tokens x 32 ch), B[k=c][n=o] = W[o][c]
    const _Float16* xrow = xt + ((size_t)b * HW_ + p0 + n) * C_;
    const _Float16* wrow = W + (size_t)(o0 + n) * C_;
    #pragma unroll
    for (int kc = 0; kc < 8; ++kc) {
      const int c0 = kc * 32;
      v16h a, bf;
      #pragma unroll
      for (int r = 0; r < 8; ++r) {
        const int ka = c0 + ((r & 3) * 2) + 8 * lg + 16 * (r >> 2);
        h2 t = *(const h2*)(xrow + ka);
        a[2 * r] = t[0]; a[2 * r + 1] = t[1];
        t = *(const h2*)(wrow + ka);
        bf[2 * r] = t[0]; bf[2 * r + 1] = t[1];
      }
      acc = wmma_f16(a, bf, acc);
    }
    const float bb    = bias[o0 + n];
    const float scale = (proj == 1) ? 1.0f : 0.0625f;   // 1/sqrt(256) folded into Q1,Q2
    _Float16* dst = (proj == 0) ? q1s : (proj == 1) ? ktok : q2s;
    #pragma unroll
    for (int r = 0; r < 8; ++r) {
      const int row = r + 8 * lg;
      dst[((size_t)b * HW_ + p0 + row) * C_ + o0 + n] = (_Float16)((acc[r] + bb) * scale);
    }
  } else {
    // D[o][p]: A[m=o][k=c] = Wv[o][c], B[k=c][n=p] = Xtok[p][c]
    const _Float16* arow = W + (size_t)(o0 + n) * C_;
    const _Float16* xrow = xt + ((size_t)b * HW_ + p0 + n) * C_;
    #pragma unroll
    for (int kc = 0; kc < 8; ++kc) {
      const int c0 = kc * 32;
      v16h a, bf;
      #pragma unroll
      for (int r = 0; r < 8; ++r) {
        const int ka = c0 + ((r & 3) * 2) + 8 * lg + 16 * (r >> 2);
        h2 t = *(const h2*)(arow + ka);
        a[2 * r] = t[0]; a[2 * r + 1] = t[1];
        t = *(const h2*)(xrow + ka);
        bf[2 * r] = t[0]; bf[2 * r + 1] = t[1];
      }
      acc = wmma_f16(a, bf, acc);
    }
    #pragma unroll
    for (int r = 0; r < 8; ++r) {
      const int row = r + 8 * lg;               // output channel within tile
      vt[(size_t)b * C_ * HW_ + (size_t)(o0 + row) * HW_ + p0 + n] =
          (_Float16)(acc[r] + bias[o0 + row]);
    }
  }
}

// -------- kernel 2: dual-softmax attention, one workgroup (4 waves) per 16-query tile.
#define PROW 1034   // row stride (halves): 1034/2=517 dwords -> conflict-free lane mapping
__global__ __launch_bounds__(128) void k_attn(
    const _Float16* __restrict__ q1s, const _Float16* __restrict__ ktok,
    const _Float16* __restrict__ q2s, const _Float16* __restrict__ vt,
    _Float16* __restrict__ h1, _Float16* __restrict__ h12) {
  __shared__ __attribute__((aligned(16))) _Float16 P1[16][PROW];   // scores -> exp(s-max), f16
  __shared__ __attribute__((aligned(16))) _Float16 P2[16][PROW];
  __shared__ float redmax[4][2][16];
  __shared__ float rowmax[2][16];
  __shared__ float rowsum[2][16];

  const int lane = threadIdx.x & 31;
  const int wid  = threadIdx.x >> 5;
  const int lg   = lane >> 4;
  const int n    = lane & 15;
  const int b    = blockIdx.x >> 6;
  const int q0   = (blockIdx.x & 63) * 16;

  if (threadIdx.x < 32) rowsum[threadIdx.x >> 4][threadIdx.x & 15] = 0.0f;

  // preload Q fragments (entire C=256 reduction: 8 chunks of 32)
  v16h qa1[8], qa2[8];
  {
    const _Float16* q1row = q1s + ((size_t)b * HW_ + q0 + n) * C_;
    const _Float16* q2row = q2s + ((size_t)b * HW_ + q0 + n) * C_;
    #pragma unroll
    for (int kc = 0; kc < 8; ++kc) {
      #pragma unroll
      for (int r = 0; r < 8; ++r) {
        const int ka = kc * 32 + ((r & 3) * 2) + 8 * lg + 16 * (r >> 2);
        h2 t = *(const h2*)(q1row + ka);
        qa1[kc][2 * r] = t[0]; qa1[kc][2 * r + 1] = t[1];
        t = *(const h2*)(q2row + ka);
        qa2[kc][2 * r] = t[0]; qa2[kc][2 * r + 1] = t[1];
      }
    }
  }

  // pass A: S1 = Q1*K^T (pre-scaled), S12 = S1 + Q2*K^T -> LDS f16; track row maxima
  float m1[8], m12[8];
  #pragma unroll
  for (int r = 0; r < 8; ++r) { m1[r] = -1e30f; m12[r] = -1e30f; }

  #pragma unroll 1
  for (int kt = wid * 16; kt < wid * 16 + 16; ++kt) {
    const int t0 = kt * 16;
    v8f s1 = {}, s2 = {};
    const _Float16* krow = ktok + ((size_t)b * HW_ + t0 + n) * C_;
    #pragma unroll
    for (int kc = 0; kc < 8; ++kc) {
      v16h bf;
      #pragma unroll
      for (int r = 0; r < 8; ++r) {
        h2 t = *(const h2*)(krow + kc * 32 + ((r & 3) * 2) + 8 * lg + 16 * (r >> 2));
        bf[2 * r] = t[0]; bf[2 * r + 1] = t[1];
      }
      s1 = wmma_f16(qa1[kc], bf, s1);
      s2 = wmma_f16(qa2[kc], bf, s2);
    }
    #pragma unroll
    for (int r = 0; r < 8; ++r) {
      const int row = r + 8 * lg;
      const float v1  = s1[r];
      const float v12 = v1 + s2[r];
      P1[row][t0 + n] = (_Float16)v1;
      P2[row][t0 + n] = (_Float16)v12;
      m1[r]  = fmaxf(m1[r],  v1);
      m12[r] = fmaxf(m12[r], v12);
    }
  }
  #pragma unroll
  for (int r = 0; r < 8; ++r) {
    #pragma unroll
    for (int s = 1; s < 16; s <<= 1) {        // reduce within each 16-lane group (wave32)
      m1[r]  = fmaxf(m1[r],  __shfl_xor(m1[r],  s, 32));
      m12[r] = fmaxf(m12[r], __shfl_xor(m12[r], s, 32));
    }
  }
  if (n == 0) {
    #pragma unroll
    for (int r = 0; r < 8; ++r) {
      redmax[wid][0][r + 8 * lg] = m1[r];
      redmax[wid][1][r + 8 * lg] = m12[r];
    }
  }
  __syncthreads();
  if (threadIdx.x < 32) {
    const int mat = threadIdx.x >> 4, row = threadIdx.x & 15;
    float m = redmax[0][mat][row];
    for (int w = 1; w < 4; ++w) m = fmaxf(m, redmax[w][mat][row]);
    rowmax[mat][row] = m;
  }
  __syncthreads();

  // pass B: exponentiate in place + row sums (normalization deferred to output scale)
  {
    const int row  = threadIdx.x >> 3;
    const int cseg = (threadIdx.x & 7) * 128;
    const float mx1 = rowmax[0][row], mx2 = rowmax[1][row];
    float s1sum = 0.0f, s2sum = 0.0f;
    for (int c = cseg; c < cseg + 128; ++c) {
      const float e1 = __expf((float)P1[row][c] - mx1);
      const float e2 = __expf((float)P2[row][c] - mx2);
      P1[row][c] = (_Float16)e1;
      P2[row][c] = (_Float16)e2;
      s1sum += e1; s2sum += e2;
    }
    atomicAdd(&rowsum[0][row], s1sum);
    atomicAdd(&rowsum[1][row], s2sum);
  }
  __syncthreads();

  // pass C: H = (P * V) / rowsum  ; V is channel-major so B-fragments are contiguous
  #pragma unroll 1
  for (int cti = 0; cti < 4; ++cti) {
    const int oc0 = (wid * 4 + cti) * 16;
    v8f a1 = {}, a2 = {};
    const _Float16* vrow = vt + (size_t)b * C_ * HW_ + (size_t)(oc0 + n) * HW_;
    #pragma unroll 1
    for (int kc = 0; kc < 32; ++kc) {
      v16h pa1, pa2, bf;
      #pragma unroll
      for (int r = 0; r < 8; ++r) {
        const int ka = kc * 32 + ((r & 3) * 2) + 8 * lg + 16 * (r >> 2);
        h2 t = *(const h2*)&P1[n][ka];
        pa1[2 * r] = t[0]; pa1[2 * r + 1] = t[1];
        t = *(const h2*)&P2[n][ka];
        pa2[2 * r] = t[0]; pa2[2 * r + 1] = t[1];
        t = *(const h2*)(vrow + ka);
        bf[2 * r] = t[0]; bf[2 * r + 1] = t[1];
      }
      a1 = wmma_f16(pa1, bf, a1);
      a2 = wmma_f16(pa2, bf, a2);
    }
    #pragma unroll
    for (int r = 0; r < 8; ++r) {
      const int row = r + 8 * lg;
      const float i1 = 1.0f / rowsum[0][row];
      const float i2 = 1.0f / rowsum[1][row];
      h1 [((size_t)b * HW_ + q0 + row) * C_ + oc0 + n] = (_Float16)(a1[r] * i1);
      h12[((size_t)b * HW_ + q0 + row) * C_ + oc0 + n] = (_Float16)(a2[r] * i2);
    }
  }
}

// -------- kernel 3: output projections + bias + residual, straight into channel-major d_out
__global__ __launch_bounds__(256) void k_outproj(
    const _Float16* __restrict__ h1, const _Float16* __restrict__ h12,
    const float* __restrict__ x1,
    const _Float16* __restrict__ wh,                    // pre-converted wps (idx 4), wpc (idx 5)
    const float* __restrict__ bps, const float* __restrict__ bpc,
    float* __restrict__ out) {
  const int lane = threadIdx.x & 31;
  const int wid  = threadIdx.x >> 5;
  const int lg   = lane >> 4;
  const int n    = lane & 15;
  int tile = blockIdx.x * 8 + wid;     // 0..65535
  const int j  = tile >> 15;           // 0: wps/h1 -> out1 ; 1: wpc/h12 -> out2
  tile &= 32767;
  const int b  = tile >> 10;
  const int o0 = ((tile >> 6) & 15) * 16;
  const int p0 = (tile & 63) * 16;

  const _Float16* ht  = j ? h12 : h1;
  const _Float16* W   = wh + (size_t)(4 + j) * WSZ;
  const float* bias   = j ? bpc : bps;
  float* dst = out + (size_t)j * B_ * C_ * HW_;

  const _Float16* arow = W + (size_t)(o0 + n) * C_;
  const _Float16* brow = ht + ((size_t)b * HW_ + p0 + n) * C_;
  v8f acc = {};
  #pragma unroll
  for (int kc = 0; kc < 8; ++kc) {
    const int c0 = kc * 32;
    v16h a, bf;
    #pragma unroll
    for (int r = 0; r < 8; ++r) {
      const int ka = c0 + ((r & 3) * 2) + 8 * lg + 16 * (r >> 2);
      h2 t = *(const h2*)(arow + ka);
      a[2 * r] = t[0]; a[2 * r + 1] = t[1];
      t = *(const h2*)(brow + ka);
      bf[2 * r] = t[0]; bf[2 * r + 1] = t[1];
    }
    acc = wmma_f16(a, bf, acc);
  }
  #pragma unroll
  for (int r = 0; r < 8; ++r) {
    const int o = o0 + r + 8 * lg;
    const size_t idx = (size_t)b * C_ * HW_ + (size_t)o * HW_ + p0 + n;
    dst[idx] = x1[idx] + bias[o] + acc[r];
  }
}

extern "C" void kernel_launch(void* const* d_in, const int* in_sizes, int n_in,
                              void* d_out, int out_size, void* d_ws, size_t ws_size,
                              hipStream_t stream) {
  (void)in_sizes; (void)n_in; (void)out_size; (void)ws_size;
  const float* x1  = (const float*)d_in[0];
  const float* x2  = (const float*)d_in[1];
  const float* wq1 = (const float*)d_in[2];
  const float* bq1 = (const float*)d_in[3];
  const float* wk  = (const float*)d_in[4];
  const float* bk  = (const float*)d_in[5];
  const float* wv  = (const float*)d_in[6];
  const float* bv  = (const float*)d_in[7];
  const float* wq2 = (const float*)d_in[8];
  const float* bq2 = (const float*)d_in[9];
  const float* wps = (const float*)d_in[10];
  const float* bps = (const float*)d_in[11];
  const float* wpc = (const float*)d_in[12];
  const float* bpc = (const float*)d_in[13];
  float* out = (float*)d_out;

  const size_t TOK = (size_t)B_ * HW_ * C_;   // 8,388,608 elems (16 MB f16 each)
  _Float16* ws  = (_Float16*)d_ws;            // needs 8*16 MB + 0.75 MB workspace
  _Float16* x1t = ws + 0 * TOK;
  _Float16* x2t = ws + 1 * TOK;
  _Float16* q1s = ws + 2 * TOK;
  _Float16* kk  = ws + 3 * TOK;
  _Float16* q2s = ws + 4 * TOK;
  _Float16* vt  = ws + 5 * TOK;
  _Float16* h1  = ws + 6 * TOK;
  _Float16* h12 = ws + 7 * TOK;
  _Float16* wh  = ws + 8 * TOK;               // 6 x 65536 f16 weights

  k_wcvt<<<dim3(6 * WSZ / 512), 256, 0, stream>>>(wq1, wk, wv, wq2, wps, wpc, wh);
  k_tokenize<<<dim3(HW_ / 32, C_ / 32, 2 * B_), 256, 0, stream>>>(x1, x2, x1t, x2t);
  k_proj<<<dim3(B_ * 64, 8), 256, 0, stream>>>(x1t, x2t, wh, bq1, bk, bv, bq2,
                                               q1s, kk, q2s, vt);
  k_attn<<<dim3(B_ * 64), 128, 0, stream>>>(q1s, kk, q2s, vt, h1, h12);
  k_outproj<<<dim3(65536 / 8), 256, 0, stream>>>(h1, h12, x1, wh, bps, bpc, out);
}
